// DDOperator_86766929314317
// MI455X (gfx1250) — matrix operand order repaired and checked by reading, not compiled, not run on hardware
//
#include <hip/hip_runtime.h>
#include <hip/hip_bf16.h>

typedef __attribute__((ext_vector_type(16))) _Float16 v16h;
typedef __attribute__((ext_vector_type(8)))  _Float16 v8h;
typedef __attribute__((ext_vector_type(8)))  float    v8f;

#define NSX 8
#define NSY 8
#define IN_C 32
#define HID 64
#define OUT_C 32
#define NCLUST 512
#define TR_STRIDE 72                   // f16 per LDS row (16B-aligned row stride: 144B)
#define BLOCK1 512                     // MLP1 block size (16 waves)
#define WAVES1 (BLOCK1 / 32)
#define BLOCK3 256                     // MLP2 block size (8 waves)
#define WAVES3 (BLOCK3 / 32)
#define PARTIAL_FLOATS (NCLUST * HID + NCLUST)   // 33280 per partial (sums + counts)
#define WS_RESERVED 262144             // frags/gsum/gcnt/pooledh region; partials after
#define MAX_PARTIALS 256

// ---------------------------------------------------------------- helpers ----
__device__ __forceinline__ float gelu_exact(float v) {
    return 0.5f * v * (1.0f + erff(v * 0.70710678118654752440f));
}

__device__ __forceinline__ v8f wmma_f16(v16h a, v16h b, v8f c) {
    // D = A(16x32 f16) x B(32x16 f16) + C(16x16 f32)
    return __builtin_amdgcn_wmma_f32_16x16x32_f16(false, a, false, b, (short)0, c,
                                                  false, false);
}

__device__ __forceinline__ int cluster_of(float cx, float cy, int batch) {
    int ix = min(max((int)(cx * (float)NSX), 0), NSX - 1);
    int iy = min(max((int)(cy * (float)NSY), 0), NSY - 1);
    return batch * (NSX * NSY) + iy * NSX + ix;
}

// ------------------------------------------------------------ weight pack ----
// B-fragment lane layout (16-bit B 32x16): lane = (n, hi); element i holds
// W[kt*32 + (i<8 ? 8*hi+i : 16+8*hi+i-8)][nt*16 + n].  28 fragments total:
//   [0..3]   Wv1 (K=32)  kt=0,      nt=0..3
//   [4..11]  Wv2 (K=64)  kt=0..1,   nt=0..3
//   [12..23] Wt1 (K=66 padded->96)  kt=0..2, nt=0..3
//   [24..27] Wt2 (K=64)  kt=0..1,   nt=0..1 (N=32)
__global__ void pack_weights_kernel(const float* __restrict__ Wv1,
                                    const float* __restrict__ Wv2,
                                    const float* __restrict__ Wt1,
                                    const float* __restrict__ Wt2,
                                    _Float16* __restrict__ frags) {
    int f    = blockIdx.x;          // 0..27
    int lane = threadIdx.x & 31;
    int n = lane & 15, hi = lane >> 4;
    const float* W; int kt, nt, K, Nw;
    if (f < 4)       { W = Wv1; kt = 0;             nt = f;            K = IN_C; Nw = HID;  }
    else if (f < 12) { W = Wv2; kt = (f - 4) >> 2;  nt = (f - 4) & 3;  K = HID;  Nw = HID;  }
    else if (f < 24) { W = Wt1; kt = (f - 12) >> 2; nt = (f - 12) & 3; K = 66;   Nw = HID;  }
    else             { W = Wt2; kt = (f - 24) >> 1; nt = (f - 24) & 1; K = HID;  Nw = OUT_C;}
    v16h v;
#pragma unroll
    for (int i = 0; i < 16; i++) {
        int k  = kt * 32 + ((i < 8) ? (8 * hi + i) : (16 + 8 * hi + (i - 8)));
        int nn = nt * 16 + n;
        float w = (k < K) ? W[k * Nw + nn] : 0.0f;
        v[i] = (_Float16)w;
    }
    ((v16h*)frags)[f * 32 + lane] = v;
}

__global__ void zero_kernel(float* __restrict__ p, int nElem) {
    int i = blockIdx.x * blockDim.x + threadIdx.x;
    if (i < nElem) p[i] = 0.0f;
}

// --------------------------------------------------- MLP1 + segment pool ----
// If `partials` != null, each block writes its private accumulator there
// (no global atomics); otherwise it atomically flushes into gsum/gcnt.
__global__ __launch_bounds__(BLOCK1) void mlp1_pool_kernel(
    const float* __restrict__ x, const float* __restrict__ sc,
    const int* __restrict__ sb, const _Float16* __restrict__ frags,
    const float* __restrict__ bv1, const float* __restrict__ bv2,
    float* __restrict__ gsum, float* __restrict__ gcnt,
    float* __restrict__ partials, int nTiles) {
    extern __shared__ char smem[];
    float*    lsum   = (float*)smem;                                   // 512*64
    float*    lcnt   = (float*)(smem + NCLUST * HID * 4);              // 512
    _Float16* trbase = (_Float16*)(smem + NCLUST * HID * 4 + NCLUST * 4);

    int tid  = threadIdx.x;
    int lane = tid & 31, wave = tid >> 5;
    int n = lane & 15, hi = lane >> 4;

    for (int i = tid; i < PARTIAL_FLOATS; i += BLOCK1) ((float*)smem)[i] = 0.0f;
    __syncthreads();

    _Float16* mytr = trbase + wave * (16 * TR_STRIDE);

    const v16h* fr = (const v16h*)frags;
    v16h bW1[4], bW2[8];
#pragma unroll
    for (int t = 0; t < 4; t++) bW1[t] = fr[t * 32 + lane];
#pragma unroll
    for (int t = 0; t < 8; t++) bW2[t] = fr[(4 + t) * 32 + lane];
    float c1b[4], c2b[4];
#pragma unroll
    for (int t = 0; t < 4; t++) { c1b[t] = bv1[t * 16 + n]; c2b[t] = bv2[t * 16 + n]; }

    int gw = blockIdx.x * WAVES1 + wave;
    int nw = gridDim.x * WAVES1;
    for (int t = gw; t < nTiles; t += nw) {
        int pt = t * 16 + n;   // each lane pair (n, n+16) handles point row n
        float cx = sc[pt * 2 + 0], cy = sc[pt * 2 + 1];
        int cl = cluster_of(cx, cy, sb[pt]);
        if (lane < 16) atomicAdd(&lcnt[cl], 1.0f);

        if (t + nw < nTiles)
            __builtin_prefetch(x + (size_t)(t + nw) * 16 * IN_C + (size_t)n * IN_C, 0, 0);

        // A fragment of x tile: lane(n,hi) holds K = 8hi..8hi+7 and 16+8hi..+7
        const float4* xr0 = (const float4*)(x + (size_t)pt * IN_C + 8 * hi);
        float4 p0 = xr0[0], p1 = xr0[1];
        const float4* xr1 = (const float4*)(x + (size_t)pt * IN_C + 16 + 8 * hi);
        float4 p2 = xr1[0], p3 = xr1[1];
        float av[16] = {p0.x, p0.y, p0.z, p0.w, p1.x, p1.y, p1.z, p1.w,
                        p2.x, p2.y, p2.z, p2.w, p3.x, p3.y, p3.z, p3.w};
        v16h a;
#pragma unroll
        for (int i = 0; i < 16; i++) a[i] = (_Float16)av[i];

        // layer 1: K=32, N=64 -> 4 WMMAs; GELU; spill row-major to LDS
#pragma unroll
        for (int nt = 0; nt < 4; nt++) {
            v8f c;
#pragma unroll
            for (int e = 0; e < 8; e++) c[e] = c1b[nt];
            c = wmma_f16(a, bW1[nt], c);
#pragma unroll
            for (int j = 0; j < 8; j++)
                mytr[(8 * hi + j) * TR_STRIDE + nt * 16 + n] = (_Float16)gelu_exact(c[j]);
        }
        asm volatile("s_wait_dscnt 0" ::: "memory");   // wave-lockstep LDS transpose

        v16h a2[2];
#pragma unroll
        for (int kt = 0; kt < 2; kt++) {
            v8h lo = *(const v8h*)(mytr + n * TR_STRIDE + kt * 32 + 8 * hi);
            v8h hh = *(const v8h*)(mytr + n * TR_STRIDE + kt * 32 + 16 + 8 * hi);
#pragma unroll
            for (int i = 0; i < 8; i++) { a2[kt][i] = lo[i]; a2[kt][8 + i] = hh[i]; }
        }

        // layer 2: K=64, N=64 -> 8 WMMAs; scatter into LDS cluster sums
#pragma unroll
        for (int nt = 0; nt < 4; nt++) {
            v8f c;
#pragma unroll
            for (int e = 0; e < 8; e++) c[e] = c2b[nt];
            c = wmma_f16(a2[0], bW2[nt], c);
            c = wmma_f16(a2[1], bW2[4 + nt], c);
#pragma unroll
            for (int j = 0; j < 8; j++) {
                int clm = __shfl(cl, 8 * hi + j, 32);
                atomicAdd(&lsum[clm * HID + nt * 16 + n], c[j]);
            }
        }
    }
    __syncthreads();
    if (partials) {
        float* dst = partials + (size_t)blockIdx.x * PARTIAL_FLOATS;
        for (int i = tid; i < PARTIAL_FLOATS; i += BLOCK1) dst[i] = ((float*)smem)[i];
    } else {
        for (int i = tid; i < NCLUST * HID; i += BLOCK1) atomicAdd(&gsum[i], lsum[i]);
        for (int i = tid; i < NCLUST; i += BLOCK1) atomicAdd(&gcnt[i], lcnt[i]);
    }
}

// ------------------------------------------------------------- finalize -----
__global__ void pool_finalize_kernel(const float* __restrict__ gsum,
                                     const float* __restrict__ gcnt,
                                     _Float16* __restrict__ pooledh) {
    int i = blockIdx.x * blockDim.x + threadIdx.x;
    if (i < NCLUST * HID) {
        float c = gcnt[i >> 6];
        pooledh[i] = (_Float16)(gsum[i] / fmaxf(c, 1.0f));
    }
}

__global__ void pool_finalize_partial_kernel(const float* __restrict__ partials,
                                             int P, _Float16* __restrict__ pooledh) {
    int i = blockIdx.x * blockDim.x + threadIdx.x;
    if (i >= NCLUST * HID) return;
    float s = 0.0f, c = 0.0f;
    for (int p = 0; p < P; p++) {
        const float* part = partials + (size_t)p * PARTIAL_FLOATS;
        s += part[i];
        c += part[NCLUST * HID + (i >> 6)];
    }
    pooledh[i] = (_Float16)(s / fmaxf(c, 1.0f));
}

// ----------------------------------------------------------------- MLP2 -----
__global__ __launch_bounds__(BLOCK3) void mlp2_kernel(
    const float* __restrict__ tc, const int* __restrict__ tb,
    const _Float16* __restrict__ frags, const _Float16* __restrict__ pooledh,
    const float* __restrict__ bt1, const float* __restrict__ bt2,
    float* __restrict__ out, int nTiles) {
    extern __shared__ char smem[];
    int tid  = threadIdx.x;
    int lane = tid & 31, wave = tid >> 5;
    int n = lane & 15, hi = lane >> 4;
    _Float16* mytr = (_Float16*)smem + wave * (16 * TR_STRIDE);

    const v16h* fr = (const v16h*)frags;
    v16h bT1[12], bT2[4];
#pragma unroll
    for (int t = 0; t < 12; t++) bT1[t] = fr[(12 + t) * 32 + lane];
#pragma unroll
    for (int t = 0; t < 4; t++)  bT2[t] = fr[(24 + t) * 32 + lane];
    float c1b[4], c2b[2];
#pragma unroll
    for (int t = 0; t < 4; t++) c1b[t] = bt1[t * 16 + n];
#pragma unroll
    for (int t = 0; t < 2; t++) c2b[t] = bt2[t * 16 + n];

    int gw = blockIdx.x * WAVES3 + wave;
    int nw = gridDim.x * WAVES3;
    for (int t = gw; t < nTiles; t += nw) {
        int pt = t * 16 + n;
        float cx = tc[pt * 2 + 0], cy = tc[pt * 2 + 1];
        int cl = cluster_of(cx, cy, tb[pt]);
        const _Float16* prow = pooledh + cl * HID;

        // feat row = [cx, cy, pooled[0..63], 0-pad to 96]; build 3 A fragments
        v16h A[3];
#pragma unroll
        for (int kt = 0; kt < 3; kt++) {
#pragma unroll
            for (int i = 0; i < 16; i++) {
                int k = kt * 32 + ((i < 8) ? (8 * hi + i) : (16 + 8 * hi + (i - 8)));
                _Float16 v;
                if (k == 0)      v = (_Float16)cx;
                else if (k == 1) v = (_Float16)cy;
                else if (k < 66) v = prow[k - 2];
                else             v = (_Float16)0.0f;
                A[kt][i] = v;
            }
        }

        // layer 1: K=96, N=64 -> 12 WMMAs
#pragma unroll
        for (int nt = 0; nt < 4; nt++) {
            v8f c;
#pragma unroll
            for (int e = 0; e < 8; e++) c[e] = c1b[nt];
            c = wmma_f16(A[0], bT1[nt], c);
            c = wmma_f16(A[1], bT1[4 + nt], c);
            c = wmma_f16(A[2], bT1[8 + nt], c);
#pragma unroll
            for (int j = 0; j < 8; j++)
                mytr[(8 * hi + j) * TR_STRIDE + nt * 16 + n] = (_Float16)gelu_exact(c[j]);
        }
        asm volatile("s_wait_dscnt 0" ::: "memory");

        v16h a2[2];
#pragma unroll
        for (int kt = 0; kt < 2; kt++) {
            v8h lo = *(const v8h*)(mytr + n * TR_STRIDE + kt * 32 + 8 * hi);
            v8h hh = *(const v8h*)(mytr + n * TR_STRIDE + kt * 32 + 16 + 8 * hi);
#pragma unroll
            for (int i = 0; i < 8; i++) { a2[kt][i] = lo[i]; a2[kt][8 + i] = hh[i]; }
        }

        // layer 2: K=64, N=32 -> 4 WMMAs; coalesced f32 stores
#pragma unroll
        for (int nt = 0; nt < 2; nt++) {
            v8f c;
#pragma unroll
            for (int e = 0; e < 8; e++) c[e] = c2b[nt];
            c = wmma_f16(a2[0], bT2[nt], c);
            c = wmma_f16(a2[1], bT2[2 + nt], c);
#pragma unroll
            for (int j = 0; j < 8; j++)
                out[(size_t)(t * 16 + 8 * hi + j) * OUT_C + nt * 16 + n] = c[j];
        }
    }
}

// ---------------------------------------------------------------- launch ----
extern "C" void kernel_launch(void* const* d_in, const int* in_sizes, int n_in,
                              void* d_out, int out_size, void* d_ws, size_t ws_size,
                              hipStream_t stream) {
    const float* x   = (const float*)d_in[0];
    const float* sc  = (const float*)d_in[1];
    const int*   sb  = (const int*)d_in[2];
    const float* tcd = (const float*)d_in[3];
    const int*   tb  = (const int*)d_in[4];
    const float* Wv1 = (const float*)d_in[5];
    const float* bv1 = (const float*)d_in[6];
    const float* Wv2 = (const float*)d_in[7];
    const float* bv2 = (const float*)d_in[8];
    const float* Wt1 = (const float*)d_in[9];
    const float* bt1 = (const float*)d_in[10];
    const float* Wt2 = (const float*)d_in[11];
    const float* bt2 = (const float*)d_in[12];
    float* out = (float*)d_out;

    int N      = in_sizes[2];      // 800000 points (divisible by 16)
    int nTiles = N / 16;

    char* ws = (char*)d_ws;
    _Float16* frags   = (_Float16*)ws;                              // 28*512*2 = 28672 B
    float*    gsum    = (float*)(ws + 32768);                       // 512*64*4 = 131072 B
    float*    gcnt    = (float*)(ws + 32768 + 131072);              // 512*4    = 2048 B
    _Float16* pooledh = (_Float16*)(ws + 32768 + 131072 + 2048);    // 512*64*2 = 65536 B
    float*    parts   = (float*)(ws + WS_RESERVED);

    // ws-size-adaptive choice of partial-buffer count (deterministic per env)
    size_t partBytes = (size_t)PARTIAL_FLOATS * sizeof(float);
    int P = 0;
    if (ws_size > WS_RESERVED) {
        size_t avail = (ws_size - WS_RESERVED) / partBytes;
        P = (int)(avail < (size_t)MAX_PARTIALS ? avail : (size_t)MAX_PARTIALS);
    }
    bool usePartials = (P >= 16);

    size_t lds1 = (size_t)NCLUST * HID * 4 + NCLUST * 4 +
                  (size_t)WAVES1 * 16 * TR_STRIDE * 2;              // 169984 B
    size_t lds3 = (size_t)WAVES3 * 16 * TR_STRIDE * 2;              // 18432 B
    (void)hipFuncSetAttribute((const void*)mlp1_pool_kernel,
                              hipFuncAttributeMaxDynamicSharedMemorySize, (int)lds1);

    pack_weights_kernel<<<28, 32, 0, stream>>>(Wv1, Wv2, Wt1, Wt2, frags);

    if (usePartials) {
        mlp1_pool_kernel<<<P, BLOCK1, lds1, stream>>>(x, sc, sb, frags, bv1, bv2,
                                                      gsum, gcnt, parts, nTiles);
        pool_finalize_partial_kernel<<<(NCLUST * HID + 255) / 256, 256, 0, stream>>>(
            parts, P, pooledh);
    } else {
        zero_kernel<<<(PARTIAL_FLOATS + 255) / 256, 256, 0, stream>>>(gsum,
                                                                      PARTIAL_FLOATS);
        mlp1_pool_kernel<<<64, BLOCK1, lds1, stream>>>(x, sc, sb, frags, bv1, bv2,
                                                       gsum, gcnt, nullptr, nTiles);
        pool_finalize_kernel<<<(NCLUST * HID + 255) / 256, 256, 0, stream>>>(
            gsum, gcnt, pooledh);
    }

    mlp2_kernel<<<1024, BLOCK3, lds3, stream>>>(tcd, tb, frags, pooledh, bt1, bt2,
                                                out, nTiles);
}
